// Seq2SeqWithTeacherForcing_15015205667194
// MI455X (gfx1250) — matrix operand order repaired and checked by reading, not compiled, not run on hardware
//
#include <hip/hip_runtime.h>
#include <hip/hip_bf16.h>

// ---------------------------------------------------------------------------
// Seq2Seq LSTM (B=64, S=512, I=H=1024) for gfx1250 (CDNA5, wave32, WMMA).
// Strategy:
//   * Convert all weights to bf16 once per call; [Wih||Whh] concatenated per
//     row -> one 4096x2048 bf16 "Wcat" per LSTM so the recurrent step is a
//     single K=2048 GEMM. 34MB of bf16 weights stay resident in the 192MB L2.
//   * One fused kernel per LSTM step: grid (4 m-tiles x 16 n-groups), 128
//     threads (4 waves). Each wave computes a 16x16 patch of H but all FOUR
//     gate tiles (i,f,g,o) -> 4 v8f accumulators, 4 v_wmma per k-iter with
//     A-operand reused 4x. Epilogue applies sigmoid/tanh cell update
//     in-register, updates c (f32, in place), writes h as bf16 into the
//     double-buffered next-step A operand [x_t || h_t], and converts the
//     next x timestep slice. Entire LSTM step = one kernel launch.
//   * Decoder adds an FC kernel per step (pred = h @ fc_W^T + fc_b) that also
//     performs the teacher-forcing select and writes the next decoder input.
// ---------------------------------------------------------------------------

typedef __attribute__((ext_vector_type(16))) __bf16 v16bf;
typedef __attribute__((ext_vector_type(8)))  __bf16 v8bf;
typedef __attribute__((ext_vector_type(8)))  float  v8f;

constexpr int Bn = 64;
constexpr int Sn = 512;
constexpr int In = 1024;
constexpr int Hn = 1024;
constexpr int KC = 2048;   // concatenated K (I + H) for the LSTM step GEMM

__device__ __forceinline__ float sigmoidf_(float x) {
    return 1.0f / (1.0f + __expf(-x));
}

// ---------------------------------------------------------------------------
// Weight prep: Wcat[n][k] = (k < In) ? Wih[n][k] : Whh[n][k-In], bf16.
// ---------------------------------------------------------------------------
__global__ void build_wcat_kernel(const float* __restrict__ Wih,
                                  const float* __restrict__ Whh,
                                  __bf16* __restrict__ dst) {
    int i = blockIdx.x * 256 + threadIdx.x;          // over 4096*2048
    int n = i >> 11;
    int k = i & (KC - 1);
    float v = (k < In) ? Wih[n * In + k] : Whh[n * Hn + (k - In)];
    dst[i] = (__bf16)v;
}

__global__ void cvt_f32_bf16_kernel(const float* __restrict__ src,
                                    __bf16* __restrict__ dst, int n) {
    int i = blockIdx.x * 256 + threadIdx.x;
    if (i < n) dst[i] = (__bf16)src[i];
}

// buf0 x-part = bf16(x[:,0,:]); buf0 h-part = 0; c = 0
__global__ void init_state_kernel(const float* __restrict__ x0,
                                  __bf16* __restrict__ buf0,
                                  float* __restrict__ c) {
    int i = blockIdx.x * 256 + threadIdx.x;          // over 64*2048
    int b = i >> 11;
    int col = i & (KC - 1);
    buf0[i] = (col < In) ? (__bf16)x0[(size_t)b * Sn * In + col] : (__bf16)0.0f;
    if (i < Bn * Hn) c[i] = 0.0f;
}

// decoder initial input: buf x-part = bf16(x[:, S-1, :])
__global__ void dec_init_kernel(const float* __restrict__ xlast,
                                __bf16* __restrict__ buf) {
    int i = blockIdx.x * 256 + threadIdx.x;          // over 64*1024
    int b = i >> 10;
    int col = i & (In - 1);
    buf[b * KC + col] = (__bf16)xlast[(size_t)b * Sn * In + col];
}

// ---------------------------------------------------------------------------
// Fused LSTM step. grid = (4, 16), block = 128 (4 waves).
//   Ain / Aout : 64 x 2048 bf16, [x_t || h_t] double buffers
//   W          : 4096 x 2048 bf16 (gate-major rows: i,f,g,o blocks of 1024)
//   bias       : 4096 f32; c : 64 x 1024 f32 (in place)
//   xnext      : f32 slice x + (t+1)*In (nullptr -> skip x conversion)
// ---------------------------------------------------------------------------
__global__ __launch_bounds__(128)
void lstm_step_kernel(const __bf16* __restrict__ Ain,
                      __bf16* __restrict__ Aout,
                      const __bf16* __restrict__ W,
                      const float* __restrict__ bias,
                      float* __restrict__ c,
                      const float* __restrict__ xnext) {
    const int lane = threadIdx.x & 31;
    const int wave = threadIdx.x >> 5;
    const int m0   = blockIdx.x * 16;                 // batch tile
    const int n0   = blockIdx.y * 64 + wave * 16;     // hidden tile
    const int nl   = lane & 15;
    const int hi   = lane >> 4;                       // K-half select

    // A operand: lane holds row m0+nl, 16 contiguous K at [kk+16*hi, +16)
    const __bf16* arow = Ain + (m0 + nl) * KC;
    const int aoff = hi * 16;

    // B operand: lane holds Wcat row (gate*H + n0 + nl), 16 contiguous K at half hi
    const __bf16* b0 = W + ((size_t)(0 * Hn + n0 + nl)) * KC + hi * 16;
    const __bf16* b1 = W + ((size_t)(1 * Hn + n0 + nl)) * KC + hi * 16;
    const __bf16* b2 = W + ((size_t)(2 * Hn + n0 + nl)) * KC + hi * 16;
    const __bf16* b3 = W + ((size_t)(3 * Hn + n0 + nl)) * KC + hi * 16;

    v8f acc0 = {}, acc1 = {}, acc2 = {}, acc3 = {};

    for (int kk = 0; kk < KC; kk += 32) {
        v16bf a = *(const v16bf*)(arow + kk + aoff);
        v16bf w0 = *(const v16bf*)(b0 + kk);
        v16bf w1 = *(const v16bf*)(b1 + kk);
        v16bf w2 = *(const v16bf*)(b2 + kk);
        v16bf w3 = *(const v16bf*)(b3 + kk);
        acc0 = __builtin_amdgcn_wmma_f32_16x16x32_bf16(false, a, false, w0,
                                                       (short)0, acc0, false, false);
        acc1 = __builtin_amdgcn_wmma_f32_16x16x32_bf16(false, a, false, w1,
                                                       (short)0, acc1, false, false);
        acc2 = __builtin_amdgcn_wmma_f32_16x16x32_bf16(false, a, false, w2,
                                                       (short)0, acc2, false, false);
        acc3 = __builtin_amdgcn_wmma_f32_16x16x32_bf16(false, a, false, w3,
                                                       (short)0, acc3, false, false);
    }

    // Cell update epilogue (C/D layout: VGPR v -> row m0+v+8*hi, col n0+nl)
    const int n = n0 + nl;
    const float bi = bias[0 * Hn + n];
    const float bf = bias[1 * Hn + n];
    const float bg = bias[2 * Hn + n];
    const float bo = bias[3 * Hn + n];
#pragma unroll
    for (int v = 0; v < 8; ++v) {
        const int m = m0 + v + hi * 8;
        float gi = acc0[v] + bi;
        float gf = acc1[v] + bf;
        float gg = acc2[v] + bg;
        float go = acc3[v] + bo;
        float co = c[m * Hn + n];
        float cn = sigmoidf_(gf) * co + sigmoidf_(gi) * tanhf(gg);
        float hn = sigmoidf_(go) * tanhf(cn);
        c[m * Hn + n] = cn;
        Aout[m * KC + In + n] = (__bf16)hn;
    }

    // Convert next x timestep slice into Aout x-part (encoder only)
    if (xnext) {
        const int cbase = blockIdx.y * 64;
#pragma unroll
        for (int j = 0; j < 8; ++j) {
            int idx = threadIdx.x * 8 + j;            // 16 rows x 64 cols
            int r = idx >> 6, cix = idx & 63;
            int b = m0 + r, col = cbase + cix;
            Aout[b * KC + col] = (__bf16)xnext[(size_t)b * Sn * In + col];
        }
    }
}

// ---------------------------------------------------------------------------
// Decoder FC + teacher forcing. grid = (4, 16), block = 128 (4 waves).
//   Hrow   : bf16 h_new rows (row stride KC, pre-offset to h-part)
//   W      : 1024 x 1024 bf16 fc weights (row-major N x K)
//   out    : pre-offset out + t*In       (index b*Sn*In + n)
//   target : pre-offset target + t*In
//   AoutX  : next decoder input x-part (row stride KC), nullptr on last step
//   maskp  : &tf_mask[t] (numpy bool -> 1 byte)
// ---------------------------------------------------------------------------
__global__ __launch_bounds__(128)
void fc_step_kernel(const __bf16* __restrict__ Hrow,
                    const __bf16* __restrict__ W,
                    const float* __restrict__ bias,
                    float* __restrict__ out,
                    const float* __restrict__ target,
                    __bf16* __restrict__ AoutX,
                    const unsigned char* __restrict__ maskp) {
    const int lane = threadIdx.x & 31;
    const int wave = threadIdx.x >> 5;
    const int m0   = blockIdx.x * 16;
    const int n0   = blockIdx.y * 64 + wave * 16;
    const int nl   = lane & 15;
    const int hi   = lane >> 4;

    const __bf16* arow = Hrow + (m0 + nl) * KC;
    const int aoff = hi * 16;
    const __bf16* brow = W + (size_t)(n0 + nl) * Hn + hi * 16;

    v8f acc = {};
    for (int kk = 0; kk < Hn; kk += 32) {
        v16bf a = *(const v16bf*)(arow + kk + aoff);
        v16bf w = *(const v16bf*)(brow + kk);
        acc = __builtin_amdgcn_wmma_f32_16x16x32_bf16(false, a, false, w,
                                                      (short)0, acc, false, false);
    }

    const int n = n0 + nl;
    const float bn = bias[n];
    const unsigned char mt = *maskp;
#pragma unroll
    for (int v = 0; v < 8; ++v) {
        const int m = m0 + v + hi * 8;
        float p = acc[v] + bn;
        out[(size_t)m * Sn * In + n] = p;
        if (AoutX) {
            float nx = mt ? target[(size_t)m * Sn * In + n] : p;
            AoutX[m * KC + n] = (__bf16)nx;
        }
    }
}

// ---------------------------------------------------------------------------
extern "C" void kernel_launch(void* const* d_in, const int* in_sizes, int n_in,
                              void* d_out, int out_size, void* d_ws, size_t ws_size,
                              hipStream_t stream) {
    const float* x        = (const float*)d_in[0];
    const float* target   = (const float*)d_in[1];
    const float* enc_Wih  = (const float*)d_in[2];
    const float* enc_Whh  = (const float*)d_in[3];
    const float* enc_b    = (const float*)d_in[4];
    const float* dec_Wih  = (const float*)d_in[5];
    const float* dec_Whh  = (const float*)d_in[6];
    const float* dec_b    = (const float*)d_in[7];
    const float* fc_W     = (const float*)d_in[8];
    const float* fc_b     = (const float*)d_in[9];
    const unsigned char* tf_mask = (const unsigned char*)d_in[10];
    float* out = (float*)d_out;

    // Workspace layout (bf16 = 2B):
    //   [0,16M)   encWcat 4096x2048 bf16
    //   [16,32M)  decWcat 4096x2048 bf16
    //   [32,34M)  fcW 1024x1024 bf16
    //   then      buf0, buf1 (64x2048 bf16 each), c (64x1024 f32)
    char* ws = (char*)d_ws;
    __bf16* encW = (__bf16*)(ws);
    __bf16* decW = (__bf16*)(ws + (size_t)16 * 1024 * 1024);
    __bf16* fcW  = (__bf16*)(ws + (size_t)32 * 1024 * 1024);
    __bf16* buf0 = (__bf16*)(ws + (size_t)34 * 1024 * 1024);
    __bf16* buf1 = buf0 + Bn * KC;
    float*  cbuf = (float*)(ws + (size_t)34 * 1024 * 1024 +
                            (size_t)2 * Bn * KC * sizeof(__bf16));
    __bf16* bufs[2] = {buf0, buf1};

    const int WCAT_ELEMS = 4 * Hn * KC;               // 8,388,608
    build_wcat_kernel<<<WCAT_ELEMS / 256, 256, 0, stream>>>(enc_Wih, enc_Whh, encW);
    build_wcat_kernel<<<WCAT_ELEMS / 256, 256, 0, stream>>>(dec_Wih, dec_Whh, decW);
    cvt_f32_bf16_kernel<<<(In * Hn + 255) / 256, 256, 0, stream>>>(fc_W, fcW, In * Hn);
    init_state_kernel<<<(Bn * KC + 255) / 256, 256, 0, stream>>>(x, buf0, cbuf);

    dim3 grid(Bn / 16, Hn / 64);                      // (4, 16)
    dim3 blk(128);

    // Encoder: 512 fused steps; final h lands (bf16) in buf0 h-part, c in cbuf.
    for (int t = 0; t < Sn; ++t) {
        const float* xn = (t + 1 < Sn) ? (x + (size_t)(t + 1) * In) : nullptr;
        lstm_step_kernel<<<grid, blk, 0, stream>>>(
            bufs[t & 1], bufs[(t + 1) & 1], encW, enc_b, cbuf, xn);
    }

    // Decoder initial input: x[:, S-1, :] into buf0 x-part (h-part = enc final h).
    dec_init_kernel<<<(Bn * In + 255) / 256, 256, 0, stream>>>(
        x + (size_t)(Sn - 1) * In, buf0);

    for (int t = 0; t < Sn; ++t) {
        __bf16* bo = bufs[(t + 1) & 1];
        lstm_step_kernel<<<grid, blk, 0, stream>>>(
            bufs[t & 1], bo, decW, dec_b, cbuf, nullptr);
        __bf16* ax = (t + 1 < Sn) ? bo : nullptr;
        fc_step_kernel<<<grid, blk, 0, stream>>>(
            bo + In, fcW, fc_b, out + (size_t)t * In,
            target + (size_t)t * In, ax, tf_mask + t);
    }
}
